// InfoNCELoss_7696581394728
// MI455X (gfx1250) — matrix operand order repaired and checked by reading, not compiled, not run on hardware
//
#include <hip/hip_runtime.h>
#include <math.h>

typedef __attribute__((ext_vector_type(16))) _Float16 v16h;
typedef __attribute__((ext_vector_type(8)))  _Float16 v8h;
typedef __attribute__((ext_vector_type(8)))  float    v8f;

#define N_ROWS   8192
#define DIM      256
#define INV_T    (1.0f / 0.07f)
#define NSLICES  16
#define TILES_PER_SLICE (N_ROWS / 16 / NSLICES)   // 32 column tiles per slice
#define ROWS_PER_WG 256                           // 8 waves x 32 rows
#define NBUF 3                                    // triple-buffered LDS tiles

// ---------------------------------------------------------------------------
// Pass 1: row-normalize u and v (fp32), emit f16 normalized copies.
// grid = (N_ROWS, 2), block = 256 (== DIM). blockIdx.y selects u or v.
// ---------------------------------------------------------------------------
__global__ __launch_bounds__(256)
void normalize_rows(const float* __restrict__ u, const float* __restrict__ v,
                    _Float16* __restrict__ uh, _Float16* __restrict__ vh) {
    __shared__ float red[256];
    const int row = blockIdx.x;
    const float*   src = blockIdx.y ? v  : u;
    _Float16*      dst = blockIdx.y ? vh : uh;

    float x = src[row * DIM + threadIdx.x];
    red[threadIdx.x] = x * x;
    __syncthreads();
    #pragma unroll
    for (int off = 128; off > 0; off >>= 1) {
        if (threadIdx.x < off) red[threadIdx.x] += red[threadIdx.x + off];
        __syncthreads();
    }
    const float scale = 1.0f / fmaxf(sqrtf(red[0]), 1e-8f);
    dst[row * DIM + threadIdx.x] = (_Float16)(x * scale);
}

// ---------------------------------------------------------------------------
// Pass 2: streaming GEMM (un @ vn^T) with fused exp-sum + diagonal capture.
// grid = (N_ROWS/256, NSLICES), block = 256 (8 waves).
// Wave w owns rows [blockIdx.x*256 + w*32, +32) as two 16-row WMMA strips.
// Column tiles (16 x 256 f16 = 8 KB) are staged into triple-buffered LDS
// with the CDNA5 async-to-LDS engine (ASYNCcnt), one barrier per tile.
// ---------------------------------------------------------------------------
__global__ __launch_bounds__(256) __attribute__((amdgpu_waves_per_eu(2, 4)))
void gemm_lse(const _Float16* __restrict__ uh, const _Float16* __restrict__ vh,
              float* __restrict__ partial,   // [NSLICES][N_ROWS] sum-of-exp
              float* __restrict__ diag) {    // [N_ROWS] scaled diagonal logit
    __shared__ __align__(32) _Float16 Bt[NBUF][16 * DIM];   // 3 x 8 KB

    const int t      = threadIdx.x;
    const int lane   = t & 31;
    const int wave   = t >> 5;
    const int half   = lane >> 4;     // 0: lanes 0-15, 1: lanes 16-31
    const int l16    = lane & 15;
    const int wbase  = blockIdx.x * ROWS_PER_WG + wave * 32; // wave's row base
    const int rowoff = half * 8;      // C/D layout: row = strip + r + 8*half

    // --- Preload A fragments: 2 strips x 8 K-steps (contiguous k order) ---
    // lane: m = l16; element e: k = 32*s + 16*half + e (contiguous)
    v16h a[2][8];
    #pragma unroll
    for (int p = 0; p < 2; ++p) {
        const _Float16* arow = uh + (size_t)(wbase + 16 * p + l16) * DIM;
        #pragma unroll
        for (int s = 0; s < 8; ++s) {
            const int k0 = 32 * s + 16 * half;
            v8h lo = *(const v8h*)(arow + k0);
            v8h hi = *(const v8h*)(arow + k0 + 8);
            v16h av;
            #pragma unroll
            for (int i = 0; i < 8; ++i) { av[i] = lo[i]; av[i + 8] = hi[i]; }
            a[p][s] = av;
        }
    }

    float esum[2][8];
    #pragma unroll
    for (int p = 0; p < 2; ++p)
        #pragma unroll
        for (int r = 0; r < 8; ++r) esum[p][r] = 0.0f;

    const int nt0   = blockIdx.y * TILES_PER_SLICE;
    const int ldRow = t >> 4;          // 0..15 : tile row this thread stages
    const int ldK   = (t & 15) * 16;   // 0..240: k offset (32 B per thread)

    // async stage: copy 32 B/thread of column tile nt into LDS buffer `buf`
    auto stage = [&](int nti, int buf) {
        unsigned ldsoff = (unsigned)(uintptr_t)(&Bt[buf][ldRow * DIM + ldK]);
        const _Float16* gsrc =
            vh + (size_t)((nt0 + nti) * 16 + ldRow) * DIM + ldK;
        asm volatile("global_load_async_to_lds_b128 %0, %1, off"
                     :: "v"(ldsoff), "v"(gsrc) : "memory");
        asm volatile("global_load_async_to_lds_b128 %0, %1, off offset:16"
                     :: "v"(ldsoff), "v"(gsrc) : "memory");
    };

    stage(0, 0);   // prologue
    for (int nti = 0; nti < TILES_PER_SLICE; ++nti) {
        const int buf = nti % NBUF;
        if (nti + 1 < TILES_PER_SLICE) {
            stage(nti + 1, (nti + 1) % NBUF);
            // in-order async completion: <=2 outstanding => tile nti landed
            asm volatile("s_wait_asynccnt 2" ::: "memory");
        } else {
            asm volatile("s_wait_asynccnt 0" ::: "memory");
        }
        __syncthreads();   // all waves' portions of tile nti visible

        // --- two 16x16 tiles, full K=256: 16 x v_wmma_f32_16x16x32_f16 ---
        // B layout: n = l16, element e: k = 32*s + 16*half + e (contiguous)
        v8f acc0 = {}, acc1 = {};
        #pragma unroll
        for (int s = 0; s < 8; ++s) {
            v16h b = *(const v16h*)(&Bt[buf][l16 * DIM + 32 * s + 16 * half]);
            acc0 = __builtin_amdgcn_wmma_f32_16x16x32_f16(
                false, a[0][s], false, b, (short)0, acc0, false, false);
            acc1 = __builtin_amdgcn_wmma_f32_16x16x32_f16(
                false, a[1][s], false, b, (short)0, acc1, false, false);
        }

        // fused softmax-denominator accumulation (no overflow: |logit|<=14.3)
        #pragma unroll
        for (int r = 0; r < 8; ++r) {
            esum[0][r] += __expf(acc0[r] * INV_T);
            esum[1][r] += __expf(acc1[r] * INV_T);
        }

        // diagonal lives in exactly one (wave-uniform) column tile per strip
        const int nt16 = (nt0 + nti) * 16;
        if (nt16 == wbase) {
            #pragma unroll
            for (int r = 0; r < 8; ++r)
                if (l16 == r + rowoff)
                    diag[wbase + r + rowoff] = acc0[r] * INV_T;
        }
        if (nt16 == wbase + 16) {
            #pragma unroll
            for (int r = 0; r < 8; ++r)
                if (l16 == r + rowoff)
                    diag[wbase + 16 + r + rowoff] = acc1[r] * INV_T;
        }
    }

    // reduce Sigma-exp across the 16-lane group sharing each row
    #pragma unroll
    for (int p = 0; p < 2; ++p) {
        #pragma unroll
        for (int r = 0; r < 8; ++r) {
            float s = esum[p][r];
            s += __shfl_xor(s, 1, 32);
            s += __shfl_xor(s, 2, 32);
            s += __shfl_xor(s, 4, 32);
            s += __shfl_xor(s, 8, 32);
            if (l16 == 0)
                partial[blockIdx.y * N_ROWS + wbase + 16 * p + r + rowoff] = s;
        }
    }
}

// ---------------------------------------------------------------------------
// Pass 3: deterministic combine: mean over rows of log(sum exp) - diag.
// single block of 256.
// ---------------------------------------------------------------------------
__global__ __launch_bounds__(256)
void finalize(const float* __restrict__ partial, const float* __restrict__ diag,
              float* __restrict__ out) {
    __shared__ float red[256];
    float acc = 0.0f;
    for (int m = threadIdx.x; m < N_ROWS; m += 256) {
        float s = 0.0f;
        #pragma unroll
        for (int p = 0; p < NSLICES; ++p) s += partial[p * N_ROWS + m];
        acc += logf(s) - diag[m];
    }
    red[threadIdx.x] = acc;
    __syncthreads();
    #pragma unroll
    for (int off = 128; off > 0; off >>= 1) {
        if (threadIdx.x < off) red[threadIdx.x] += red[threadIdx.x + off];
        __syncthreads();
    }
    if (threadIdx.x == 0) out[0] = red[0] / (float)N_ROWS;
}

// ---------------------------------------------------------------------------
extern "C" void kernel_launch(void* const* d_in, const int* in_sizes, int n_in,
                              void* d_out, int out_size, void* d_ws, size_t ws_size,
                              hipStream_t stream) {
    (void)in_sizes; (void)n_in; (void)out_size; (void)ws_size;
    const float* u = (const float*)d_in[0];
    const float* v = (const float*)d_in[1];
    float* out = (float*)d_out;

    // workspace: uh[8192*256] f16 | vh[8192*256] f16 |
    //            partial[16][8192] f32 | diag[8192] f32   (~9 MB)
    _Float16* uh = (_Float16*)d_ws;
    _Float16* vh = uh + (size_t)N_ROWS * DIM;
    float* partial = (float*)(vh + (size_t)N_ROWS * DIM);
    float* diag    = partial + NSLICES * N_ROWS;

    normalize_rows<<<dim3(N_ROWS, 2), 256, 0, stream>>>(u, v, uh, vh);
    gemm_lse<<<dim3(N_ROWS / ROWS_PER_WG, NSLICES), 256, 0, stream>>>(
        uh, vh, partial, diag);
    finalize<<<1, 256, 0, stream>>>(partial, diag, out);
}
